// MyRNN_35021163331655
// MI455X (gfx1250) — matrix operand order repaired and checked by reading, not compiled, not run on hardware
//
#include <hip/hip_runtime.h>
#include <hip/hip_fp16.h>

// ---------------------------------------------------------------------------
// MI455X (gfx1250, wave32) stacked-LSTM pipeline.
//  * Big GEMMs + recurrence use v_wmma_f32_16x16x32_f16 (f16 in, f32 acc).
//  * gx activations stored as f16 in WMMA-accumulator-native tile order
//    (halves the dominant HBM traffic term: 2 GB -> 1 GB).
//  * Weights pre-packed into WMMA B-fragment order (lane-major 32B/lane).
//  * Recurrence: w_hh held entirely in VGPRs, h in LDS, c in accumulators;
//    next-step gx prefetched (global_prefetch_b8) during WMMA compute.
//  * GEMM A-tile staged with global_load_async_to_lds_b128 (ASYNCcnt).
// ---------------------------------------------------------------------------

typedef __attribute__((ext_vector_type(16))) _Float16     v16h;
typedef __attribute__((ext_vector_type(8)))  _Float16     v8h;
typedef __attribute__((ext_vector_type(8)))  float        v8f;
typedef __attribute__((ext_vector_type(4)))  unsigned int v4u;
typedef __attribute__((ext_vector_type(4)))  float        v4f;

union Frag16 { v16h h; v4u u[2]; };   // one WMMA f16 A/B fragment per lane (32B)
union AccU   { v8f  f; v4f q[2]; };   // one WMMA f32 C/D fragment per lane (32B)
union H8     { v8h  h; v4u u;    };   // 8 halves = 16B (f16 gx tile slice)

__device__ __forceinline__ float sigf(float x) { return 1.0f / (1.0f + __expf(-x)); }

// ---------------------------------------------------------------------------
// Pack a row-major (N x Kin) f32 weight into WMMA-B fragment order (f16):
// out[kt][nt][lane*16 + ki], h=lane/16: ki<8 -> K=kt*32+8h+ki, ki>=8 -> K=kt*32+16+8h+(ki-8),
// col = nt*16 + lane%16. K zero-padded to Kp.
// ---------------------------------------------------------------------------
__global__ void pack_b_kernel(const float* __restrict__ w, _Float16* __restrict__ out,
                              int N, int Kin, int Kp) {
  const int ntiles = N >> 4;
  const int total  = (Kp >> 5) * ntiles * 512;
  for (int idx = blockIdx.x * blockDim.x + threadIdx.x; idx < total;
       idx += gridDim.x * blockDim.x) {
    int e    = idx & 511;
    int nt   = (idx >> 9) % ntiles;
    int kt   = idx / (512 * ntiles);
    int lane = e >> 4;
    int ki   = e & 15;
    int col  = nt * 16 + (lane & 15);
    int k    = kt * 32 + ((ki >> 3) << 4) + ((lane >> 4) << 3) + (ki & 7);
    float v  = (k < Kin) ? w[(size_t)col * Kin + k] : 0.0f;
    out[idx] = (_Float16)v;
  }
}

__global__ void bias_add_kernel(const float* __restrict__ a, const float* __restrict__ b,
                                float* __restrict__ o, int n) {
  int i = blockIdx.x * blockDim.x + threadIdx.x;
  if (i < n) o[i] = a[i] + b[i];
}

// ---------------------------------------------------------------------------
// x (2048,120,60,3) f32 -> inpH [t'][b][kp] f16, kp in [0,192) (180 padded).
// Matches jnp.transpose(x,(0,3,1,2)).reshape(-1,120,180):
//   f = t'*180+kp ; c=f/7200 ; t=(f%7200)/60 ; w=f%60 ; value = x[b,t,w,c].
// Stored t-major so M-tiles of the GEMM are 16 consecutive batch rows at one t.
// ---------------------------------------------------------------------------
__global__ void prep_x_kernel(const float* __restrict__ x, _Float16* __restrict__ inpH) {
  const long long total = 120LL * 2048 * 192;
  for (long long idx = (long long)blockIdx.x * blockDim.x + threadIdx.x; idx < total;
       idx += (long long)gridDim.x * blockDim.x) {
    int kp = (int)(idx % 192);
    int b  = (int)((idx / 192) % 2048);
    int tp = (int)(idx / (192LL * 2048));
    float v = 0.0f;
    if (kp < 180) {
      int f = tp * 180 + kp;
      int c = f / 7200;
      int r = f % 7200;
      int t = r / 60;
      int w = r % 60;
      v = x[(((size_t)b * 120 + t) * 60 + w) * 3 + c];
    }
    inpH[idx] = (_Float16)v;
  }
}

// ---------------------------------------------------------------------------
// gx = A (MxK f16 row-major) @ Bp^T + bias, N = 512 fixed.
// Block = 8 waves; block owns one 16-row M-tile, waves split N (64 cols each).
// A tile staged to LDS with async-to-LDS loads; B read coalesced from packed
// buffer (L2-resident). Output f16 in accumulator tile order:
// gxp[mt][nt][lane][8] halves (one b128 store per tile per lane).
// ---------------------------------------------------------------------------
__global__ void gemm_gx_kernel(const _Float16* __restrict__ A,
                               const _Float16* __restrict__ Bp,
                               const float* __restrict__ bias,
                               _Float16* __restrict__ gxp, int K) {
  __shared__ _Float16 sA[16 * 192];
  const int tid  = threadIdx.x;
  const int wave = tid >> 5;
  const int lane = tid & 31;
  const size_t mt = blockIdx.x;

  {
    // Stage the 16xK A-tile into LDS via the CDNA5 async datapath
    // (GLOBAL_LOAD_ASYNC_TO_LDS_B128, tracked by ASYNCcnt).
    const _Float16* gbase = A + mt * 16 * (size_t)K;
    const unsigned  lbase = (unsigned)(size_t)(&sA[0]);
    for (int i = tid * 8; i < 16 * K; i += 256 * 8) {
      unsigned long long gaddr = (unsigned long long)(const void*)(gbase + i);
      unsigned           laddr = lbase + (unsigned)(i * 2);
      asm volatile("global_load_async_to_lds_b128 %0, %1, off"
                   :: "v"(laddr), "v"(gaddr)
                   : "memory");
    }
    asm volatile("s_wait_asynccnt 0x0" ::: "memory");
  }
  __syncthreads();

  AccU acc[4];
#pragma unroll
  for (int j = 0; j < 4; ++j) {
    float bv = bias[wave * 64 + j * 16 + (lane & 15)];
#pragma unroll
    for (int r = 0; r < 8; ++r) acc[j].f[r] = bv;
  }

  const int row  = lane & 15;
  const int koff = (lane >> 4) << 3;  // A-frag: lanes 0-15 hold K 0..7/16..23, 16-31 hold 8..15/24..31
  const int nk   = K >> 5;
  for (int k = 0; k < nk; ++k) {
    Frag16 af;
    const _Float16* ap = sA + row * K + k * 32 + koff;
    af.u[0] = *(const v4u*)ap;
    af.u[1] = *(const v4u*)(ap + 16);
#pragma unroll
    for (int j = 0; j < 4; ++j) {
      const int nt = wave * 4 + j;
      const _Float16* wp = Bp + ((size_t)k * 32 + nt) * 512 + lane * 16;
      Frag16 bf;
      bf.u[0] = *(const v4u*)wp;
      bf.u[1] = *(const v4u*)(wp + 8);
      acc[j].f = __builtin_amdgcn_wmma_f32_16x16x32_f16(
          false, af.h, false, bf.h, (short)0, acc[j].f, false, false);
    }
  }

#pragma unroll
  for (int j = 0; j < 4; ++j) {
    const int nt = wave * 4 + j;
    H8 o;
#pragma unroll
    for (int r = 0; r < 8; ++r) o.h[r] = (_Float16)acc[j].f[r];
    _Float16* gp = gxp + ((mt * 32 + nt) * 32 + lane) * 8;
    *(v4u*)gp = o.u;
  }
}

// ---------------------------------------------------------------------------
// LSTM recurrence, H=128 fixed. Block = 16 batch rows, 8 waves.
// Wave w owns hidden units [w*16, w*16+16): gate n-tiles {w, 8+w, 16+w, 24+w}.
// w_hh fragments (16 x 32B) live in VGPRs for the whole 120-step loop.
// h (16x128 f16) in LDS; c-state in f32 registers. gx for step t+1 is
// prefetched while step t's WMMAs run. Writes h_t for all t as f16 [t][B][128].
// ---------------------------------------------------------------------------
__global__ void lstm_scan_kernel(const _Float16* __restrict__ gxp,
                                 const _Float16* __restrict__ whhp,
                                 _Float16* __restrict__ hOut,
                                 int T, int Btot) {
  __shared__ _Float16 sH[16 * 128];
  const int tid  = threadIdx.x;
  const int wave = tid >> 5;
  const int lane = tid & 31;
  const int b0   = blockIdx.x * 16;

  for (int i = tid; i < 16 * 128; i += 256) sH[i] = (_Float16)0.0f;

  Frag16 bf[4][4];  // [ktile][gate], entirely register-resident
#pragma unroll
  for (int k = 0; k < 4; ++k)
#pragma unroll
    for (int g = 0; g < 4; ++g) {
      const _Float16* wp = whhp + ((size_t)k * 32 + (g * 8 + wave)) * 512 + lane * 16;
      bf[k][g].u[0] = *(const v4u*)wp;
      bf[k][g].u[1] = *(const v4u*)(wp + 8);
    }

  v8f c_f = {};
  __syncthreads();

  const int row  = lane & 15;
  const int koff = (lane >> 4) << 3;
  const int n    = wave * 16 + row;     // hidden-unit column this lane produces
  const int mb   = (lane >> 4) * 8;     // first batch row of this lane's 8 elems
  const size_t tstep = (size_t)(Btot / 16) * 32 * 32 * 8;  // gx halves per timestep

  for (int t = 0; t < T; ++t) {
    const size_t mt = (size_t)t * (Btot / 16) + blockIdx.x;
    AccU acc[4];
#pragma unroll
    for (int g = 0; g < 4; ++g) {       // init with precomputed gx (incl. biases)
      const _Float16* gp = gxp + ((mt * 32 + (g * 8 + wave)) * 32 + lane) * 8;
      H8 gv;
      gv.u = *(const v4u*)gp;
#pragma unroll
      for (int r = 0; r < 8; ++r) acc[g].f[r] = (float)gv.h[r];
      if (t + 1 < T) __builtin_prefetch(gp + tstep, 0, 0);  // -> global_prefetch_b8
    }
#pragma unroll
    for (int k = 0; k < 4; ++k) {       // gates += h_prev @ w_hh^T
      Frag16 af;
      const _Float16* ap = sH + row * 128 + k * 32 + koff;
      af.u[0] = *(const v4u*)ap;
      af.u[1] = *(const v4u*)(ap + 16);
#pragma unroll
      for (int g = 0; g < 4; ++g)
        acc[g].f = __builtin_amdgcn_wmma_f32_16x16x32_f16(
            false, af.h, false, bf[k][g].h, (short)0, acc[g].f, false, false);
    }
    // i/f/g/o tiles are elementwise-aligned in the accumulator layout.
    float hv[8];
#pragma unroll
    for (int r = 0; r < 8; ++r) {
      float cv = sigf(acc[1].f[r]) * c_f[r] + sigf(acc[0].f[r]) * tanhf(acc[2].f[r]);
      c_f[r] = cv;
      hv[r]  = sigf(acc[3].f[r]) * tanhf(cv);
    }
    __syncthreads();  // everyone done reading sH(t-1)
#pragma unroll
    for (int r = 0; r < 8; ++r) {
      _Float16 hh = (_Float16)hv[r];
      sH[(mb + r) * 128 + n] = hh;
      hOut[((size_t)t * Btot + b0 + mb + r) * 128 + n] = hh;
    }
    __syncthreads();  // sH(t) visible before next step's reads
  }
}

// ---------------------------------------------------------------------------
// lstm2 (4 layers, H=10, T=4) + 40x10 output head: one thread per batch elem.
// Input = lstm1 last-step h (b,128) viewed as (4,32).
// ---------------------------------------------------------------------------
struct L2Params {
  const float* wih0; const float* whh0; const float* bih0; const float* bhh0;
  const float* wih1; const float* whh1; const float* bih1; const float* bhh1;
  const float* wih2; const float* whh2; const float* bih2; const float* bhh2;
  const float* wih3; const float* whh3; const float* bih3; const float* bhh3;
  const float* ow;   const float* ob;
};

__global__ void lstm2_out_kernel(const _Float16* __restrict__ hLast,
                                 L2Params p, float* __restrict__ out, int B) {
  int b = blockIdx.x * blockDim.x + threadIdx.x;
  if (b >= B) return;
  const _Float16* xr = hLast + (size_t)b * 128;

  float seq[4][10];
  {  // layer 0: input dim 32
    float h[10], c[10];
#pragma unroll
    for (int j = 0; j < 10; ++j) { h[j] = 0.0f; c[j] = 0.0f; }
#pragma unroll
    for (int t = 0; t < 4; ++t) {
      float xv[32];
#pragma unroll
      for (int k = 0; k < 32; ++k) xv[k] = (float)xr[t * 32 + k];
      float gt[40];
#pragma unroll
      for (int j = 0; j < 40; ++j) {
        float a = p.bih0[j] + p.bhh0[j];
        const float* wi = p.wih0 + j * 32;
#pragma unroll
        for (int k = 0; k < 32; ++k) a += wi[k] * xv[k];
        const float* wh = p.whh0 + j * 10;
#pragma unroll
        for (int k = 0; k < 10; ++k) a += wh[k] * h[k];
        gt[j] = a;
      }
#pragma unroll
      for (int j = 0; j < 10; ++j) {
        float cv = sigf(gt[10 + j]) * c[j] + sigf(gt[j]) * tanhf(gt[20 + j]);
        c[j] = cv;
        h[j] = sigf(gt[30 + j]) * tanhf(cv);
        seq[t][j] = h[j];
      }
    }
  }
  const float* wih[3] = { p.wih1, p.wih2, p.wih3 };
  const float* whh[3] = { p.whh1, p.whh2, p.whh3 };
  const float* bih[3] = { p.bih1, p.bih2, p.bih3 };
  const float* bhh[3] = { p.bhh1, p.bhh2, p.bhh3 };
  for (int L = 0; L < 3; ++L) {  // layers 1..3: input dim 10
    float h[10], c[10];
#pragma unroll
    for (int j = 0; j < 10; ++j) { h[j] = 0.0f; c[j] = 0.0f; }
#pragma unroll
    for (int t = 0; t < 4; ++t) {
      float xv[10];
#pragma unroll
      for (int k = 0; k < 10; ++k) xv[k] = seq[t][k];
      float gt[40];
#pragma unroll
      for (int j = 0; j < 40; ++j) {
        float a = bih[L][j] + bhh[L][j];
        const float* wi = wih[L] + j * 10;
        const float* wh = whh[L] + j * 10;
#pragma unroll
        for (int k = 0; k < 10; ++k) a += wi[k] * xv[k] + wh[k] * h[k];
        gt[j] = a;
      }
#pragma unroll
      for (int j = 0; j < 10; ++j) {
        float cv = sigf(gt[10 + j]) * c[j] + sigf(gt[j]) * tanhf(gt[20 + j]);
        c[j] = cv;
        h[j] = sigf(gt[30 + j]) * tanhf(cv);
        seq[t][j] = h[j];
      }
    }
  }
  float* op = out + (size_t)b * 40;
#pragma unroll
  for (int j = 0; j < 40; ++j) {
    float a = p.ob[j];
    const float* wo = p.ow + j * 10;
#pragma unroll
    for (int k = 0; k < 10; ++k) a += wo[k] * seq[3][k];
    op[j] = a;
  }
}

// ---------------------------------------------------------------------------
// Input order (setup_inputs dict, nested lists flattened in order):
//  [0] x ; [1..4] lstm1 L0 (w_ih,w_hh,b_ih,b_hh) ; [5..8] lstm1 L1 ;
//  [9..24] lstm2 L0..L3 (4 each) ; [25] out_w ; [26] out_b.
// Output: (2048,4,10) f32 = 81920 floats.
// ---------------------------------------------------------------------------
extern "C" void kernel_launch(void* const* d_in, const int* in_sizes, int n_in,
                              void* d_out, int out_size, void* d_ws, size_t ws_size,
                              hipStream_t stream) {
  const float* x     = (const float*)d_in[0];
  const float* wih1a = (const float*)d_in[1];
  const float* whh1a = (const float*)d_in[2];
  const float* bih1a = (const float*)d_in[3];
  const float* bhh1a = (const float*)d_in[4];
  const float* wih1b = (const float*)d_in[5];
  const float* whh1b = (const float*)d_in[6];
  const float* bih1b = (const float*)d_in[7];
  const float* bhh1b = (const float*)d_in[8];

  L2Params p;
  p.wih0 = (const float*)d_in[9];  p.whh0 = (const float*)d_in[10];
  p.bih0 = (const float*)d_in[11]; p.bhh0 = (const float*)d_in[12];
  p.wih1 = (const float*)d_in[13]; p.whh1 = (const float*)d_in[14];
  p.bih1 = (const float*)d_in[15]; p.bhh1 = (const float*)d_in[16];
  p.wih2 = (const float*)d_in[17]; p.whh2 = (const float*)d_in[18];
  p.bih2 = (const float*)d_in[19]; p.bhh2 = (const float*)d_in[20];
  p.wih3 = (const float*)d_in[21]; p.whh3 = (const float*)d_in[22];
  p.bih3 = (const float*)d_in[23]; p.bhh3 = (const float*)d_in[24];
  p.ow   = (const float*)d_in[25]; p.ob   = (const float*)d_in[26];

  char* ws = (char*)d_ws;
  size_t off = 0;
  auto take = [&](size_t bytes) -> char* {
    off = (off + 255) & ~(size_t)255;
    char* r = ws + off;
    off += bytes;
    return r;
  };

  _Float16* inpH  = (_Float16*)take(120ull * 2048 * 192 * 2);  // 94 MB
  _Float16* hAll1 = (_Float16*)take(120ull * 2048 * 128 * 2);  // 63 MB
  _Float16* hAll2 = (_Float16*)take(120ull * 2048 * 128 * 2);  // 63 MB
  _Float16* gx    = (_Float16*)take(245760ull * 512 * 2);      // 251 MB f16 (reused both layers)
  _Float16* wih1p = (_Float16*)take(6ull  * 32 * 512 * 2);
  _Float16* wih2p = (_Float16*)take(4ull  * 32 * 512 * 2);
  _Float16* whh1p = (_Float16*)take(4ull  * 32 * 512 * 2);
  _Float16* whh2p = (_Float16*)take(4ull  * 32 * 512 * 2);
  float*    bias1 = (float*)   take(512 * 4);
  float*    bias2 = (float*)   take(512 * 4);

  // Pack weights / fold biases (tiny).
  pack_b_kernel<<<96, 256, 0, stream>>>(wih1a, wih1p, 512, 180, 192);
  pack_b_kernel<<<64, 256, 0, stream>>>(wih1b, wih2p, 512, 128, 128);
  pack_b_kernel<<<64, 256, 0, stream>>>(whh1a, whh1p, 512, 128, 128);
  pack_b_kernel<<<64, 256, 0, stream>>>(whh1b, whh2p, 512, 128, 128);
  bias_add_kernel<<<2, 256, 0, stream>>>(bih1a, bhh1a, bias1, 512);
  bias_add_kernel<<<2, 256, 0, stream>>>(bih1b, bhh1b, bias2, 512);

  // Reshape/transpose + f16 conversion, t-major layout.
  prep_x_kernel<<<4096, 256, 0, stream>>>(x, inpH);

  // LSTM1 layer 0: gx GEMM (M=245760, K=192, N=512) then 120-step scan.
  gemm_gx_kernel<<<15360, 256, 0, stream>>>(inpH, wih1p, bias1, gx, 192);
  lstm_scan_kernel<<<128, 256, 0, stream>>>(gx, whh1p, hAll1, 120, 2048);

  // LSTM1 layer 1: K=128.
  gemm_gx_kernel<<<15360, 256, 0, stream>>>(hAll1, wih2p, bias2, gx, 128);
  lstm_scan_kernel<<<128, 256, 0, stream>>>(gx, whh2p, hAll2, 120, 2048);

  // LSTM2 stack + output head on last-step h of lstm1.
  lstm2_out_kernel<<<8, 256, 0, stream>>>(hAll2 + 119ull * 2048 * 128, p,
                                          (float*)d_out, 2048);

  (void)in_sizes; (void)n_in; (void)out_size; (void)ws_size;
}
